// CNN_66795331387590
// MI455X (gfx1250) — matrix operand order repaired and checked by reading, not compiled, not run on hardware
//
#include <hip/hip_runtime.h>

// ---------------------------------------------------------------------------
// SE(3)-CNN forward for MI455X (gfx1250, wave32, WMMA).
// Convs run as implicit GEMM on v_wmma_f32_16x16x32_f16 with:
//   - zero-padded NDHWC activations (channel-last, channels padded to 32x):
//       * no spatial bounds checks (halo of 3 zero voxels)
//       * no channel guards (zero-filled channel tail)
//       * A fragment = two contiguous 16B loads per lane
//   - tap-major K ordering (kd/kh/kw counter loops; no div/mod in hot loop)
//   - 4 m-tiles per wave sharing each B fragment (4x less B traffic)
// ---------------------------------------------------------------------------

typedef __attribute__((ext_vector_type(16))) _Float16 v16h;
typedef __attribute__((ext_vector_type(8)))  _Float16 v8h;
typedef __attribute__((ext_vector_type(8)))  float    v8f;

#define KT   343      // 7^3 taps
#define NANG 16
#define NRAD 3
#define PAD  3
#define MBLK 4        // m-tiles per wave

// ---------------------------------------------------------------------------
// 1) Basis: RADIAL[3][343], ANGULAR[16][343]
// ---------------------------------------------------------------------------
__global__ void k_basis(float* __restrict__ rad, float* __restrict__ ang) {
  int t = blockIdx.x * blockDim.x + threadIdx.x;
  if (t >= KT) return;
  int i = t / 49, j = (t / 7) % 7, kk = t % 7;
  float X = (float)i - 3.f, Y = (float)j - 3.f, Z = (float)kk - 3.f;
  float r  = sqrtf(X * X + Y * Y + Z * Z);
  float rs = r > 0.f ? r : 1.f;
  float ux = X / rs, uy = Y / rs, uz = Z / rs;
  float a[NANG] = {1.f, ux, uy, uz, ux * uy, uy * uz, uz * ux,
                   ux * ux - uy * uy, 3.f * uz * uz - 1.f,
                   ux * ux * ux, uy * uy * uy, uz * uz * uz,
                   ux * uy * uz, ux * uy * uy, uz * ux * ux, uy * uz * uz};
#pragma unroll
  for (int q = 0; q < NANG; ++q) ang[q * KT + t] = a[q];
  float mask = (r <= 3.5f) ? 1.f : 0.f;
  rad[0 * KT + t] = __expf(-(r - 0.0f) * (r - 0.0f)) * mask;
  rad[1 * KT + t] = __expf(-(r - 1.5f) * (r - 1.5f)) * mask;
  rad[2 * KT + t] = __expf(-(r - 3.0f) * (r - 3.0f)) * mask;
}

// ---------------------------------------------------------------------------
// 2) Build WMMA-swizzled f16 weight matrix, tap-major K: k = t*cinPad + ci.
//    B element (k, n): kt=k/32, half=(k%32)>>4, j=k&15, lane=half*16+(n&15)
//    flat = ((kt*nTiles + n/16)*32 + lane)*16 + j
// ---------------------------------------------------------------------------
__global__ void k_build_w(const float* __restrict__ W, const float* __restrict__ rad,
                          const float* __restrict__ ang, _Float16* __restrict__ kmat,
                          int cin, int cinPad, int cout, int nTiles) {
  long idx   = (long)blockIdx.x * blockDim.x + threadIdx.x;
  int  Npad  = nTiles * 16;
  long total = (long)KT * cinPad * Npad;
  if (idx >= total) return;
  int n = (int)(idx % Npad);
  int k = (int)(idx / Npad);
  int t  = k / cinPad;
  int ci = k % cinPad;
  float val = 0.f;
  if (n < cout && ci < cin) {
    const float* w = W + (long)(n * cin + ci) * (NRAD * NANG);
#pragma unroll
    for (int r = 0; r < NRAD; ++r) {
      float rv = rad[r * KT + t];
#pragma unroll
      for (int a = 0; a < NANG; ++a) val += w[r * NANG + a] * rv * ang[a * KT + t];
    }
  }
  int kt = k >> 5, kl = k & 31;
  int half = kl >> 4, j = kl & 15;
  int nt = n >> 4, nl = n & 15;
  long out = (((long)kt * nTiles + nt) * 32 + (half * 16 + nl)) * 16 + j;
  kmat[out] = (_Float16)val;
}

// ---------------------------------------------------------------------------
// 3) Extend: copy x then append tensor-square channels (9*b)   [NCDHW f32]
// ---------------------------------------------------------------------------
__global__ void k_ext(const float* __restrict__ x, float* __restrict__ ext,
                      int a, int b, int c, int S) {
  int base = a + 3 * b + 5 * c;
  int Cext = base + 9 * b;
  long total = 2L * Cext * S;
  long idx = (long)blockIdx.x * blockDim.x + threadIdx.x;
  if (idx >= total) return;
  int  s  = (int)(idx % S);
  long r2 = idx / S;
  int  ch = (int)(r2 % Cext);
  int  bt = (int)(r2 / Cext);
  const float* xb = x + (long)bt * base * S;
  float v;
  if (ch < base) {
    v = xb[(long)ch * S + s];
  } else {
    int q = ch - base, bi = q / 9, ij = q % 9, i = ij / 3, j = ij % 3;
    float vi = xb[(long)(a + bi * 3 + i) * S + s];
    float vj = xb[(long)(a + bi * 3 + j) * S + s];
    v = vi * vj;
  }
  ext[idx] = v;
}

// ---------------------------------------------------------------------------
// 4) SE(3) BN stats: one block per multiplicity channel; rs = rsqrt(n2+eps)
// ---------------------------------------------------------------------------
__global__ void k_bn_stats(const float* __restrict__ ext, float* __restrict__ rs,
                           int a, int b, int c, int S) {
  int mi   = blockIdx.x;
  int Cext = a + 3 * b + 5 * c + 9 * b;
  int ch0, d;
  if      (mi < a)             { ch0 = mi;                                d = 1; }
  else if (mi < a + b)         { ch0 = a + (mi - a) * 3;                  d = 3; }
  else if (mi < a + b + c)     { ch0 = a + 3 * b + (mi - a - b) * 5;      d = 5; }
  else                         { ch0 = a + 3 * b + 5 * c + (mi - a - b - c) * 9; d = 9; }
  long n = 2L * d * S;
  float acc = 0.f;
  for (long i = threadIdx.x; i < n; i += blockDim.x) {
    int  s  = (int)(i % S);
    long r2 = i / S;
    int  dd = (int)(r2 % d);
    int  bt = (int)(r2 / d);
    float v = ext[((long)bt * Cext + ch0 + dd) * S + s];
    acc += v * v;
  }
  __shared__ float red[256];
  red[threadIdx.x] = acc;
  __syncthreads();
  for (int st = 128; st > 0; st >>= 1) {
    if (threadIdx.x < st) red[threadIdx.x] += red[threadIdx.x + st];
    __syncthreads();
  }
  if (threadIdx.x == 0) {
    float n2 = red[0] / (2.f * (float)S);
    rs[mi] = rsqrtf(n2 + 1e-5f);
  }
}

// ---------------------------------------------------------------------------
// 5) BN apply + f32 -> f16 into ZERO-PADDED NDHWC volume:
//    h[bt][pz][py][px][ci] with pz/py/px in [0,Dp), ci in [0,cinPad).
//    Border voxels AND the channel tail [cin,cinPad) are written 0, so the
//    conv gather needs no bounds checks and no channel guards.
// ---------------------------------------------------------------------------
__global__ void k_bn_apply_pad(const float* __restrict__ ext, const float* __restrict__ rs,
                               _Float16* __restrict__ h, int a, int b, int c,
                               int cinPad, int Din, int Dp) {
  int Cext = a + 3 * b + 5 * c + 9 * b;
  long total = 2L * Dp * Dp * Dp * cinPad;
  long idx = (long)blockIdx.x * blockDim.x + threadIdx.x;
  if (idx >= total) return;
  int  ci  = (int)(idx % cinPad);
  long vox = idx / cinPad;
  int  pw  = (int)(vox % Dp);
  int  ph  = (int)((vox / Dp) % Dp);
  int  pd  = (int)((vox / (Dp * Dp)) % Dp);
  int  bt  = (int)(vox / ((long)Dp * Dp * Dp));
  int id = pd - PAD, ih = ph - PAD, iw = pw - PAD;
  _Float16 v = (_Float16)0.f;
  if (ci < Cext &&
      (unsigned)id < (unsigned)Din && (unsigned)ih < (unsigned)Din &&
      (unsigned)iw < (unsigned)Din) {
    int mi;
    if      (ci < a)                 mi = ci;
    else if (ci < a + 3 * b)         mi = a + (ci - a) / 3;
    else if (ci < a + 3 * b + 5 * c) mi = a + b + (ci - a - 3 * b) / 5;
    else                             mi = a + b + c + (ci - a - 3 * b - 5 * c) / 9;
    int S = Din * Din * Din;
    float x = ext[((long)bt * Cext + ci) * S + (id * Din + ih) * Din + iw];
    v = (_Float16)(x * rs[mi]);
  }
  h[idx] = v;
}

// ---------------------------------------------------------------------------
// 6) Implicit-GEMM conv: wave handles MBLK consecutive 16x16 tiles (same nt).
//    K loop = 343 taps (counter loops) x ceil(cin/32) chunks.
//    NDHWC input: each A fragment = two contiguous 16B loads per lane.
// ---------------------------------------------------------------------------
__global__ void __launch_bounds__(256)
k_conv_wmma(const _Float16* __restrict__ actp, const _Float16* __restrict__ kmat,
            const float* __restrict__ bias, float* __restrict__ out,
            int cinPad, int cout, int nTiles,
            int Dp, int Dout, int stride, int ao) {
  const int lane = threadIdx.x & 31;
  const int wave = threadIdx.x >> 5;
  const int S = Dout * Dout * Dout;
  const int mTiles  = (2 * S) >> 4;                 // 16000/16 = 1000 exact
  const int mGroups = (mTiles + MBLK - 1) / MBLK;
  long wid = (long)blockIdx.x * 8 + wave;
  if (wid >= (long)mGroups * nTiles) return;        // wave-uniform: EXEC all-1 at WMMA
  const int nt = (int)(wid % nTiles);
  const int mg = (int)(wid / nTiles);

  const int half = lane >> 4;      // K half-wave split (A/B layouts)
  const int mrow = lane & 15;      // A-matrix row held by this lane
  const int DP2 = Dp * Dp;

  // Per-lane base (in v8h units): NDHWC addr = (((bt*Dp+z)*Dp+y)*Dp+x)*cinPad
  int base8[MBLK];
#pragma unroll
  for (int mb = 0; mb < MBLK; ++mb) {
    int vox = (mg * MBLK + mb) * 16 + mrow;
    if (vox >= 2 * S) vox = 2 * S - 1;              // clamp; rows never stored
    int bt = vox / S, s = vox % S;
    int od = s / (Dout * Dout), oh = (s / Dout) % Dout, ow = s % Dout;
    int addr = (((bt * Dp + od * stride) * Dp + oh * stride) * Dp + ow * stride) * cinPad;
    base8[mb] = addr >> 3;                          // cinPad multiple of 32 -> /8 exact
  }

  const int ccTiles = cinPad >> 5;
  v8f acc[MBLK];
#pragma unroll
  for (int mb = 0; mb < MBLK; ++mb) acc[mb] = (v8f){};

  const v8h* __restrict__ a8 = (const v8h*)actp;
  int kchunk = 0;
  for (int kd = 0; kd < 7; ++kd) {
    for (int kh = 0; kh < 7; ++kh) {
      for (int kw = 0; kw < 7; ++kw) {
        // tap offset in v8h units
        const int tap8 = ((kd * DP2 + kh * Dp + kw) * cinPad) >> 3;
        for (int cc = 0; cc < ccTiles; ++cc, ++kchunk) {
          // B fragment: one contiguous 32B load per lane (pre-swizzled)
          const v16h* bp = (const v16h*)kmat + ((long)kchunk * nTiles + nt) * 32 + lane;
          v16h bf = *bp;
          __builtin_prefetch((const void*)(bp + (long)nTiles * 32), 0, 3);

          // This lane's K values: ci0+16*half+[0..16)  (A layout)
          const int c8 = ((cc << 5) + (half << 4)) >> 3;   // ci0 / 8
#pragma unroll
          for (int mb = 0; mb < MBLK; ++mb) {
            const v8h* ap = a8 + base8[mb] + tap8;
            v8h lo = ap[c8];
            v8h hi = ap[c8 + 1];
            v16h af = __builtin_shufflevector(lo, hi, 0, 1, 2, 3, 4, 5, 6, 7,
                                              8, 9, 10, 11, 12, 13, 14, 15);
            acc[mb] = __builtin_amdgcn_wmma_f32_16x16x32_f16(
                false, af, false, bf, (short)0, acc[mb], false, false);
          }
        }
      }
    }
  }

  // Epilogue: C/D layout — VGPR jj: lanes 0-15 hold M=jj, lanes 16-31 M=jj+8
  const int n = nt * 16 + (lane & 15);
#pragma unroll
  for (int mb = 0; mb < MBLK; ++mb) {
#pragma unroll
    for (int jj = 0; jj < 8; ++jj) {
      int vox2 = (mg * MBLK + mb) * 16 + jj + half * 8;
      if (vox2 < 2 * S && n < cout) {
        float v = acc[mb][jj];
        if (n < ao) v = fmaxf(v + bias[n], 0.f);
        int bt2 = vox2 / S, s2 = vox2 % S;
        out[((long)bt2 * cout + n) * S + s2] = v;
      }
    }
  }
}

// ---------------------------------------------------------------------------
// 7) Head: global mean pool * alpha * 5  +  (sum(inp) @ lin_w^T + lin_b)
// ---------------------------------------------------------------------------
__global__ void k_head(const float* __restrict__ xfin, const float* __restrict__ inp,
                       const float* __restrict__ alpha, const float* __restrict__ lin_w,
                       const float* __restrict__ lin_b, float* __restrict__ out,
                       int S, int Sin) {
  int bt = blockIdx.x;  // batch 0..1
  float acc = 0.f, lin = 0.f;
  for (int s = threadIdx.x; s < S; s += blockDim.x) acc += xfin[(long)bt * S + s];
  for (int i = threadIdx.x; i < 5 * Sin; i += blockDim.x) {
    int ch = i / Sin;
    lin += inp[((long)bt * 5 + ch) * Sin + (i % Sin)] * lin_w[ch];
  }
  __shared__ float r1[256], r2[256];
  r1[threadIdx.x] = acc; r2[threadIdx.x] = lin;
  __syncthreads();
  for (int st = 128; st > 0; st >>= 1) {
    if (threadIdx.x < st) {
      r1[threadIdx.x] += r1[threadIdx.x + st];
      r2[threadIdx.x] += r2[threadIdx.x + st];
    }
    __syncthreads();
  }
  if (threadIdx.x == 0)
    out[bt] = (r1[0] / (float)S) * alpha[0] * 5.f + r2[0] + lin_b[0];
}

// ---------------------------------------------------------------------------
// Host: deterministic pipeline on `stream`, scratch carved from d_ws.
// ---------------------------------------------------------------------------
extern "C" void kernel_launch(void* const* d_in, const int* in_sizes, int n_in,
                              void* d_out, int out_size, void* d_ws, size_t ws_size,
                              hipStream_t stream) {
  (void)in_sizes; (void)n_in; (void)out_size; (void)ws_size;

  const float* inp   = (const float*)d_in[0];
  const float* W[6]  = {(const float*)d_in[1], (const float*)d_in[2], (const float*)d_in[3],
                        (const float*)d_in[4], (const float*)d_in[5], (const float*)d_in[6]};
  const float* bs[6] = {(const float*)d_in[7], (const float*)d_in[8], (const float*)d_in[9],
                        (const float*)d_in[10], (const float*)d_in[11], nullptr};
  const float* alpha = (const float*)d_in[12];
  const float* lin_w = (const float*)d_in[13];
  const float* lin_b = (const float*)d_in[14];

  static const int A_[7] = {5, 10, 10, 16, 16, 16, 1};
  static const int B_[7] = {0, 3, 3, 8, 8, 8, 0};
  static const int C_[7] = {0, 0, 1, 1, 1, 1, 0};
  static const int strides[6] = {2, 1, 1, 1, 1, 1};
  static const int relus[6]   = {1, 1, 1, 1, 1, 0};

  int cin[6], cinPad[6], cout[6], nTiles[6], ao[6], Din[6], Dp[6];
  for (int i = 0; i < 6; ++i) {
    cin[i]    = A_[i] + 3 * B_[i] + 5 * C_[i] + 9 * B_[i];
    cinPad[i] = (cin[i] + 31) & ~31;
    cout[i]   = A_[i + 1] + 3 * B_[i + 1] + 5 * C_[i + 1];
    nTiles[i] = (cout[i] + 15) / 16;
    ao[i]     = relus[i] ? A_[i + 1] : 0;
    Din[i]    = (i == 0) ? 40 : 20;
    Dp[i]     = Din[i] + 2 * PAD;
  }
  const int Dout = 20, Sout = Dout * Dout * Dout;

  // -------- workspace carving (256B aligned) --------
  char* p = (char*)d_ws;
  auto carve = [&](size_t bytes) -> void* {
    void* r = (void*)p;
    p += (bytes + 255) & ~(size_t)255;
    return r;
  };
  float*    rad  = (float*)carve(NRAD * KT * sizeof(float));
  float*    angb = (float*)carve(NANG * KT * sizeof(float));
  _Float16* kmat[6];
  for (int i = 0; i < 6; ++i)
    kmat[i] = (_Float16*)carve((size_t)KT * cinPad[i] * nTiles[i] * 16 * sizeof(_Float16));
  float*    ext  = (float*)carve(2000000 * sizeof(float));       // max 2*117*8000
  _Float16* hbuf = (_Float16*)carve(6300000 * sizeof(_Float16)); // max 2*46^3*32
  float*    bufX = (float*)carve(800000 * sizeof(float));        // max 2*45*8000
  float*    bufY = (float*)carve(800000 * sizeof(float));
  float*    rsb  = (float*)carve(256 * sizeof(float));

  // -------- 1) basis + 2) synthesized, swizzled weights --------
  k_basis<<<(KT + 255) / 256, 256, 0, stream>>>(rad, angb);
  for (int i = 0; i < 6; ++i) {
    long total = (long)KT * cinPad[i] * nTiles[i] * 16;
    k_build_w<<<(unsigned)((total + 255) / 256), 256, 0, stream>>>(
        W[i], rad, angb, kmat[i], cin[i], cinPad[i], cout[i], nTiles[i]);
  }

  // -------- block pipeline --------
  const float* cur = inp;
  float* dst = bufX;
  float* alt = bufY;
  for (int i = 0; i < 6; ++i) {
    int a = A_[i], b = B_[i], c = C_[i];
    int Sin = Din[i] * Din[i] * Din[i];
    long extN = 2L * cin[i] * Sin;
    long padN = 2L * cinPad[i] * Dp[i] * Dp[i] * Dp[i];

    k_ext<<<(unsigned)((extN + 255) / 256), 256, 0, stream>>>(cur, ext, a, b, c, Sin);
    k_bn_stats<<<(unsigned)(a + b + c + b), 256, 0, stream>>>(ext, rsb, a, b, c, Sin);
    k_bn_apply_pad<<<(unsigned)((padN + 255) / 256), 256, 0, stream>>>(
        ext, rsb, hbuf, a, b, c, cinPad[i], Din[i], Dp[i]);

    int mGroups = ((2 * Sout / 16) + MBLK - 1) / MBLK;
    long waves = (long)mGroups * nTiles[i];
    k_conv_wmma<<<(unsigned)((waves + 7) / 8), 256, 0, stream>>>(
        hbuf, kmat[i], bs[i], dst,
        cinPad[i], cout[i], nTiles[i],
        Dp[i], Dout, strides[i], ao[i]);

    cur = dst;
    float* t = dst; dst = alt; alt = t;
  }

  // -------- head --------
  k_head<<<2, 256, 0, stream>>>(cur, inp, alpha, lin_w, lin_b, (float*)d_out,
                                Sout, 40 * 40 * 40);
}
